// MultiHeadAttention_39779987096222
// MI455X (gfx1250) — matrix operand order repaired and checked by reading, not compiled, not run on hardware
//
#include <hip/hip_runtime.h>
#include <hip/hip_bf16.h>

// Problem constants (from reference)
constexpr int SEQ = 384;
constexpr int DM  = 768;
constexpr int NHEADS = 12;
constexpr int HD  = 64;
constexpr int SSTR = SEQ + 4;    // LDS score-row stride (bank de-alias)
constexpr int CHUNK = 64;        // key rows per TDM chunk
constexpr int NCH = SEQ / CHUNK; // 6 chunks per query row
constexpr int RKSTR = HD + 4;    // 68: LDS stride produced by TDM pad feature

typedef __attribute__((ext_vector_type(16))) __bf16 v16bf;
typedef __attribute__((ext_vector_type(8)))  float  v8f;
typedef __attribute__((ext_vector_type(4)))  unsigned int u32x4;
typedef __attribute__((ext_vector_type(8)))  unsigned int u32x8;

__device__ __forceinline__ v16bf pack16(float4 a, float4 b, float4 c, float4 d) {
  v16bf r;
  r[0]  = (__bf16)a.x; r[1]  = (__bf16)a.y; r[2]  = (__bf16)a.z; r[3]  = (__bf16)a.w;
  r[4]  = (__bf16)b.x; r[5]  = (__bf16)b.y; r[6]  = (__bf16)b.z; r[7]  = (__bf16)b.w;
  r[8]  = (__bf16)c.x; r[9]  = (__bf16)c.y; r[10] = (__bf16)c.z; r[11] = (__bf16)c.w;
  r[12] = (__bf16)d.x; r[13] = (__bf16)d.y; r[14] = (__bf16)d.z; r[15] = (__bf16)d.w;
  return r;
}

// ---------------------------------------------------------------------------
// TDM: DMA a [CHUNK x 64] fp32 tile (row stride 768 floats in memory) into LDS
// with 4-DWORD padding after every 64 DWORDs -> LDS row stride 68 floats.
// D# groups per CDNA5 ISA §8.3/8.4; 2-D tensor -> groups 2/3 omitted.
// ---------------------------------------------------------------------------
__device__ __forceinline__ void tdm_load_tile(const float* gsrc, unsigned lds_off) {
  const unsigned long long ga = (unsigned long long)(uintptr_t)gsrc;
  u32x4 g0;
  g0[0] = 1u;                                                // count=1, user D#
  g0[1] = lds_off;                                           // lds_addr (bytes)
  g0[2] = (unsigned)(ga & 0xFFFFFFFFu);                      // global_addr[31:0]
  g0[3] = (unsigned)((ga >> 32) & 0x01FFFFFFu) | (2u << 30); // addr[56:32] | type=2
  u32x8 g1;
  g1[0] = (2u << 16)      // data_size = 4 bytes
        | (1u << 20)      // pad_enable
        | (5u << 22)      // pad_interval: every 64 DWORDs
        | (3u << 25);     // pad_amount: 4 DWORDs
  g1[1] = (64u << 16);             // tensor_dim0 = 64 elements   (bits 79:48 lo)
  g1[2] = ((unsigned)CHUNK << 16); // tensor_dim1 = CHUNK rows    (bits 111:80 lo)
  g1[3] = (64u << 16);             // tile_dim0 = 64              (bits 127:112)
  g1[4] = (unsigned)CHUNK;         // tile_dim1 = CHUNK, tile_dim2 = 0
  g1[5] = (unsigned)DM;            // tensor_dim0_stride = 768    (bits 207:160 lo)
  g1[6] = 0u;
  g1[7] = 0u;
  asm volatile("tensor_load_to_lds %0, %1" :: "s"(g0), "s"(g1) : "memory");
}

// ---------------------------------------------------------------------------
// Kernel 1: fused Q/K/V projection.  Out[s,o] = sum_d X[s,d]*W[o,d] + b[o]
// One 16x16 output tile per wave, v_wmma_f32_16x16x32_bf16, K = 768 = 24 steps.
// X (1.2 MB) and W (2.4 MB each) are L2-resident, so tile re-reads are cheap.
// ---------------------------------------------------------------------------
__global__ void qkv_proj_kernel(const float* __restrict__ X,
                                const float* __restrict__ Wq, const float* __restrict__ bq,
                                const float* __restrict__ Wk, const float* __restrict__ bk,
                                const float* __restrict__ Wv, const float* __restrict__ bv,
                                float* __restrict__ Qo, float* __restrict__ Ko,
                                float* __restrict__ Vo) {
  const int lane = threadIdx.x & 31;
  const int wid  = blockIdx.x * (blockDim.x >> 5) + (threadIdx.x >> 5);
  constexpr int MT = SEQ / 16;   // 24
  constexpr int NT = DM  / 16;   // 48
  if (wid >= 3 * MT * NT) return;

  const int mat = wid / (MT * NT);
  const int rem = wid % (MT * NT);
  const int m0  = (rem / NT) * 16;
  const int n0  = (rem % NT) * 16;

  const float* W; const float* bias; float* O;
  if (mat == 0)      { W = Wq; bias = bq; O = Qo; }
  else if (mat == 1) { W = Wk; bias = bk; O = Ko; }
  else               { W = Wv; bias = bv; O = Vo; }

  const int r  = lane & 15;
  const int hi = lane >> 4;

  const float bval = bias[n0 + r];   // bias depends only on the output column
  v8f acc;
  #pragma unroll
  for (int i = 0; i < 8; ++i) acc[i] = bval;

  const float* arow = X + (size_t)(m0 + r) * DM;   // A row m = lane&15
  const float* brow = W + (size_t)(n0 + r) * DM;   // B[k,n] = W[n,k]

  for (int kk = 0; kk < DM; kk += 32) {
    const int ab = kk + hi * 8;      // A half-lane K groups {kb..kb+7, kb+16..kb+23}
    const float4* ap  = (const float4*)(arow + ab);
    const float4* ap2 = (const float4*)(arow + ab + 16);
    v16bf a = pack16(ap[0], ap[1], ap2[0], ap2[1]);
    const float4* bp = (const float4*)(brow + kk + hi * 8); // B: K groups {kb..kb+7, kb+16..kb+23}
    v16bf b = pack16(bp[0], bp[1], bp[4], bp[5]);
    acc = __builtin_amdgcn_wmma_f32_16x16x32_bf16(false, a, false, b,
                                                  (short)0, acc, false, false);
  }

  float* orow = O + (size_t)(m0 + hi * 8) * DM + n0 + r;
  #pragma unroll
  for (int v = 0; v < 8; ++v) orow[(size_t)v * DM] = acc[v];
}

// ---------------------------------------------------------------------------
// Kernel 2: fused attention.  One (head, 16-query tile) per 256-thread block.
//   phase 1: scores = (Q Kᵀ)/8 + mask            -- WMMA bf16
//   phase 2: scores += (Q · RK[q])/8             -- TDM double-buffered stream
//   phase 3: row softmax                         -- wave32 shuffles
//   phase 4: ctx = probs · (V + RV[q])           -- coalesced float2 stream
// ---------------------------------------------------------------------------
__global__ void attn_kernel(const float* __restrict__ Qm, const float* __restrict__ Km,
                            const float* __restrict__ Vm,
                            const float* __restrict__ RK, const float* __restrict__ RV,
                            const float* __restrict__ mask, float* __restrict__ out) {
  const int h   = blockIdx.x % NHEADS;
  const int q0  = (blockIdx.x / NHEADS) * 16;
  const int tid = threadIdx.x;
  const int lane = tid & 31;
  const int w    = tid >> 5;                 // 8 waves of 32 (wave32)

  __shared__ float qs[16][HD];               // 4 KB
  __shared__ float sc[16][SSTR];             // 24.8 KB
  __shared__ float rkbuf[2][CHUNK][RKSTR];   // 34.8 KB TDM double buffer

  constexpr int NSTEP = 16 * NCH;            // 96 DMA chunks per block
  const float* rk_head = RK + (size_t)h * HD;

  // Kick off the first two DMA chunks; they stream while we do QKV load + WMMA.
  if (w == 0) {
    tdm_load_tile(rk_head + (size_t)q0 * SEQ * DM,
                  (unsigned)(uintptr_t)&rkbuf[0][0][0]);
    tdm_load_tile(rk_head + ((size_t)q0 * SEQ + CHUNK) * DM,
                  (unsigned)(uintptr_t)&rkbuf[1][0][0]);
  }

  // --- load Q tile (16 x 64) into LDS, 4 floats per thread -----------------
  {
    const int idx = tid * 4;                 // 256*4 = 1024 = 16*64
    const int qi = idx >> 6, d = idx & 63;
    *(float4*)&qs[qi][d] =
        *(const float4*)(Qm + (size_t)(q0 + qi) * DM + h * HD + d);
  }
  __syncthreads();

  const int r  = lane & 15;
  const int hi = lane >> 4;

  // --- phase 1: base QK^T, 24 key tiles, 3 per wave ------------------------
  for (int i = 0; i < 3; ++i) {
    const int k0 = (w + i * 8) * 16;
    v8f acc;
    #pragma unroll
    for (int j = 0; j < 8; ++j) acc[j] = 0.f;
    #pragma unroll
    for (int kk = 0; kk < HD; kk += 32) {
      const float* aq = &qs[r][kk + hi * 8];
      v16bf a = pack16(*(const float4*)(aq),      *(const float4*)(aq + 4),
                       *(const float4*)(aq + 16), *(const float4*)(aq + 20));
      const float4* bp =
          (const float4*)(Km + (size_t)(k0 + r) * DM + h * HD + kk + hi * 8);
      v16bf b = pack16(bp[0], bp[1], bp[4], bp[5]);
      acc = __builtin_amdgcn_wmma_f32_16x16x32_bf16(false, a, false, b,
                                                    (short)0, acc, false, false);
    }
    const int kcol = k0 + r;
    const float mterm = (1.0f - mask[kcol]) * -10000.0f;
    #pragma unroll
    for (int v = 0; v < 8; ++v) sc[v + hi * 8][kcol] = acc[v] * 0.125f + mterm;
  }
  __syncthreads();

  // --- phase 2: relative-key term via TDM double buffering -----------------
  // step s -> (q = s/NCH, chunk c = s%NCH); 4 lanes share one key row
  // (16 d's each, ds_read_b128, LDS stride 68 kills bank aliasing),
  // 2 shfl_xor steps reduce the 4 partials.
  {
    const int kgrp = tid >> 2;               // 0..63: key row within chunk
    const int dq   = (tid & 3) * 16;         // d-quarter
    for (int s = 0; s < NSTEP; ++s) {
      if (w == 0) {
        if (s + 1 < NSTEP) __builtin_amdgcn_s_wait_tensorcnt(1); // chunk s done
        else               __builtin_amdgcn_s_wait_tensorcnt(0);
      }
      __syncthreads();                       // LDS chunk visible to all waves
      const int q = s / NCH, c = s % NCH;
      const float4* qv = (const float4*)&qs[q][dq];
      const float4 qa = qv[0], qb = qv[1], qc = qv[2], qd = qv[3];
      const float4* rp = (const float4*)&rkbuf[s & 1][kgrp][dq];
      const float4 r0 = rp[0], r1 = rp[1], r2 = rp[2], r3 = rp[3];
      float acc = r0.x*qa.x + r0.y*qa.y + r0.z*qa.z + r0.w*qa.w
                + r1.x*qb.x + r1.y*qb.y + r1.z*qb.z + r1.w*qb.w
                + r2.x*qc.x + r2.y*qc.y + r2.z*qc.z + r2.w*qc.w
                + r3.x*qd.x + r3.y*qd.y + r3.z*qd.z + r3.w*qd.w;
      acc += __shfl_xor(acc, 1);
      acc += __shfl_xor(acc, 2);
      if ((tid & 3) == 0) sc[q][c * CHUNK + kgrp] += acc * 0.125f;
      __syncthreads();                       // buffer free before re-DMA
      if (w == 0 && s + 2 < NSTEP) {
        const int s2 = s + 2;
        const int q2 = s2 / NCH, c2 = s2 % NCH;
        tdm_load_tile(rk_head + ((size_t)(q0 + q2) * SEQ + c2 * CHUNK) * DM,
                      (unsigned)(uintptr_t)&rkbuf[s & 1][0][0]);
      }
    }
  }
  __syncthreads();

  // --- phase 3: softmax, one wave per 2 rows -------------------------------
  for (int rr = 0; rr < 2; ++rr) {
    const int row = w * 2 + rr;
    float m = -3.4e38f;
    for (int k = lane; k < SEQ; k += 32) m = fmaxf(m, sc[row][k]);
    #pragma unroll
    for (int off = 16; off > 0; off >>= 1) m = fmaxf(m, __shfl_xor(m, off));
    float sum = 0.f;
    for (int k = lane; k < SEQ; k += 32) {
      const float e = __expf(sc[row][k] - m);
      sc[row][k] = e;
      sum += e;
    }
    #pragma unroll
    for (int off = 16; off > 0; off >>= 1) sum += __shfl_xor(sum, off);
    const float inv = 1.0f / sum;
    for (int k = lane; k < SEQ; k += 32) sc[row][k] *= inv;
  }
  __syncthreads();

  // --- phase 4: ctx = probs . (V + RV[q]).  One wave per 2 rows; lane owns a
  // float2 of the 64-wide head dim -> fully coalesced 256B-per-key streams.
  for (int rr = 0; rr < 2; ++rr) {
    const int row = w * 2 + rr;
    const int d0  = lane * 2;
    const float* vp  = Vm + h * HD + d0;
    const float* rvp = RV + ((size_t)(q0 + row) * SEQ) * DM + h * HD + d0;
    float2 acc; acc.x = 0.f; acc.y = 0.f;
    #pragma unroll 4
    for (int k = 0; k < SEQ; ++k) {
      const float  p  = sc[row][k];
      const float2 vv = *(const float2*)(vp  + (size_t)k * DM);
      const float2 rv = *(const float2*)(rvp + (size_t)k * DM);
      acc.x += p * (vv.x + rv.x);
      acc.y += p * (vv.y + rv.y);
    }
    *(float2*)(out + (size_t)(q0 + row) * DM + h * HD + d0) = acc;
  }
}

extern "C" void kernel_launch(void* const* d_in, const int* in_sizes, int n_in,
                              void* d_out, int out_size, void* d_ws, size_t ws_size,
                              hipStream_t stream) {
  const float* hs   = (const float*)d_in[0];
  const float* mask = (const float*)d_in[1];
  const float* rk   = (const float*)d_in[2];
  const float* rv   = (const float*)d_in[3];
  const float* Wq   = (const float*)d_in[4];
  const float* bq   = (const float*)d_in[5];
  const float* Wk   = (const float*)d_in[6];
  const float* bk   = (const float*)d_in[7];
  const float* Wv   = (const float*)d_in[8];
  const float* bv   = (const float*)d_in[9];
  float* out = (float*)d_out;

  float* Q = (float*)d_ws;           // 3 x 384*768 fp32 = 3.5 MB scratch
  float* K = Q + SEQ * DM;
  float* V = K + SEQ * DM;

  // 3 matrices x 24 x 48 tiles = 3456 wave-jobs, 8 waves/block -> 432 blocks
  qkv_proj_kernel<<<432, 256, 0, stream>>>(hs, Wq, bq, Wk, bk, Wv, bv, Q, K, V);
  // 12 heads x 24 query tiles
  attn_kernel<<<NHEADS * (SEQ / 16), 256, 0, stream>>>(Q, K, V, rk, rv, mask, out);
}
